// OverlapRowParallelLayer_17815524344010
// MI455X (gfx1250) — matrix operand order repaired and checked by reading, not compiled, not run on hardware
//
#include <hip/hip_runtime.h>

typedef __attribute__((ext_vector_type(16))) _Float16 v16h;
typedef __attribute__((ext_vector_type(8)))  _Float16 v8h;
typedef __attribute__((ext_vector_type(8)))  float    v8f;

constexpr int WSZ = 4, MD = 4096, KD = 4096, ND = 4096;
constexpr int BM = 128, BN = 256, BK = 64;
constexpr int LDA = BK + 8;                // 72 halves -> 144B row stride (16B aligned, bank-spread)
constexpr int LDS_A = BM * LDA;            // halves per A buffer
constexpr int LDS_B = BN * LDA;            // halves per B buffer
constexpr int KSTEPS_PER_W = KD / BK;      // 64
constexpr int STAGES = WSZ * KSTEPS_PER_W; // 256

// 8 wave32 per block, arranged 2 (M) x 4 (N); each wave owns a 64x64 output patch
// = 4x4 WMMA tiles = 128 f32 accumulator VGPRs.

__global__ __launch_bounds__(256) void rowpar_allreduce_gemm_wmma(
    const _Float16* __restrict__ X,   // [W, M, K] fp16, K-major
    const _Float16* __restrict__ Wt,  // [W, N, K] fp16, K-major
    _Float16* __restrict__ Out)       // [M, N] fp16
{
  extern __shared__ _Float16 smem[];
  _Float16* sA = smem;                 // [2][BM][LDA]
  _Float16* sB = smem + 2 * LDS_A;     // [2][BN][LDA]

  const int tid   = threadIdx.x;
  const int lane  = tid & 31;
  const int wid   = tid >> 5;
  const int waveM = wid >> 2;          // 0..1
  const int waveN = wid & 3;           // 0..3
  const int lrow  = lane & 15;         // row/col within 16x16 tile
  const int lhi   = lane >> 4;         // half-wave select (K interleave)

  const int bx = blockIdx.x;           // N tile index (16 tiles)
  const int by = blockIdx.y;           // M tile index (32 tiles)

  // Uniform block bases (byte pointers) -> compiler keeps these in SGPRs.
  const char* Xblk = (const char*)X  + (size_t)by * BM * KD * 2;
  const char* Wblk = (const char*)Wt + (size_t)bx * BN * KD * 2;

  // Per-thread 32-bit staging offsets, computed once (VALU-free inner loop).
  unsigned offG_A[4], offS_A[4];       // A: 128x64 halves = 1024 chunks / 256 thr = 4
  unsigned offG_B[8], offS_B[8];       // B: 256x64 halves = 2048 chunks / 256 thr = 8
#pragma unroll
  for (int i = 0; i < 4; ++i) {
    const int c = tid + i * 256, r = c >> 3, col = (c & 7) << 3;
    offG_A[i] = (unsigned)(r * KD + col) * 2u;       // bytes in global slab
    offS_A[i] = (unsigned)(r * LDA + col);           // halves in LDS buffer
  }
#pragma unroll
  for (int i = 0; i < 8; ++i) {
    const int c = tid + i * 256, r = c >> 3, col = (c & 7) << 3;
    offG_B[i] = (unsigned)(r * KD + col) * 2u;
    offS_B[i] = (unsigned)(r * LDA + col);
  }

  v8f acc[4][4];
  const v8f zf = {0.f, 0.f, 0.f, 0.f, 0.f, 0.f, 0.f, 0.f};
#pragma unroll
  for (int i = 0; i < 4; ++i)
#pragma unroll
    for (int j = 0; j < 4; ++j) acc[i][j] = zf;

  // ---- global -> register staging: uniform stage base + u32 thread offset ----
  auto stage_gload = [&](int s, v8h* ra, v8h* rb) {
    const int w  = s >> 6;                                   // KSTEPS_PER_W == 64
    const int kb = (s & (KSTEPS_PER_W - 1)) * BK;
    const char* gA = Xblk + ((size_t)w * MD * KD + kb) * 2;  // uniform (SGPR)
    const char* gB = Wblk + ((size_t)w * ND * KD + kb) * 2;  // uniform (SGPR)
#pragma unroll
    for (int i = 0; i < 4; ++i) ra[i] = *(const v8h*)(gA + offG_A[i]);
#pragma unroll
    for (int i = 0; i < 8; ++i) rb[i] = *(const v8h*)(gB + offG_B[i]);
  };
  auto stage_sstore = [&](int buf, const v8h* ra, const v8h* rb) {
    _Float16* dA = sA + buf * LDS_A;
    _Float16* dB = sB + buf * LDS_B;
#pragma unroll
    for (int i = 0; i < 4; ++i) *(v8h*)(dA + offS_A[i]) = ra[i];
#pragma unroll
    for (int i = 0; i < 8; ++i) *(v8h*)(dB + offS_B[i]) = rb[i];
  };

  // ---- WMMA compute: 2 K-steps x (4x4) tiles = 32 WMMAs per stage ----
  auto compute = [&](int buf) {
    const _Float16* A = sA + buf * LDS_A;
    const _Float16* B = sB + buf * LDS_B;
#pragma unroll
    for (int kk = 0; kk < BK; kk += 32) {
      v16h af[4], bf[4];
#pragma unroll
      for (int i = 0; i < 4; ++i) {
        // A frag: lane<16 -> K {kk+0..7, kk+16..23}; lane>=16 -> K {kk+8..15, kk+24..31}
        const _Float16* p = A + (waveM * 64 + i * 16 + lrow) * LDA + kk + lhi * 8;
        v8h lo = *(const v8h*)p;
        v8h hi = *(const v8h*)(p + 16);
        af[i] = __builtin_shufflevector(lo, hi, 0,1,2,3,4,5,6,7,8,9,10,11,12,13,14,15);
      }
#pragma unroll
      for (int j = 0; j < 4; ++j) {
        // B frag: lane<16 -> K {kk+0..7, kk+16..23} of weight row n; lane>=16 -> K {kk+8..15, kk+24..31}
        const _Float16* p = B + (waveN * 64 + j * 16 + lrow) * LDA + kk + lhi * 8;
        v8h lo = *(const v8h*)p;
        v8h hi = *(const v8h*)(p + 16);
        bf[j] = __builtin_shufflevector(lo, hi, 0,1,2,3,4,5,6,7,8,9,10,11,12,13,14,15);
      }
#pragma unroll
      for (int i = 0; i < 4; ++i)
#pragma unroll
        for (int j = 0; j < 4; ++j)
          acc[i][j] = __builtin_amdgcn_wmma_f32_16x16x32_f16(
              false, af[i], false, bf[j], (short)0, acc[i][j], false, false);
    }
  };

  // ---- double-buffered main loop over (w, k) ----
  v8h ra[4], rb[8];
  stage_gload(0, ra, rb);
  stage_sstore(0, ra, rb);
  __syncthreads();

  for (int s = 0; s < STAGES; ++s) {
    const int buf   = s & 1;
    const bool more = (s + 1) < STAGES;
    if (more) stage_gload(s + 1, ra, rb);   // HBM loads in flight under WMMAs
    compute(buf);
    if (more) stage_sstore(buf ^ 1, ra, rb);
    __syncthreads();
  }

  // ---- epilogue: f32 acc -> f16, lane-coalesced row stores ----
#pragma unroll
  for (int i = 0; i < 4; ++i) {
#pragma unroll
    for (int j = 0; j < 4; ++j) {
      const int m0 = by * BM + waveM * 64 + i * 16 + lhi * 8; // VGPR r -> row m0+r
      const int n  = bx * BN + waveN * 64 + j * 16 + lrow;    // lanes 0..15 -> 16 consecutive cols
#pragma unroll
      for (int r = 0; r < 8; ++r)
        Out[(size_t)(m0 + r) * ND + n] = (_Float16)acc[i][j][r];
    }
  }
}

extern "C" void kernel_launch(void* const* d_in, const int* in_sizes, int n_in,
                              void* d_out, int out_size, void* d_ws, size_t ws_size,
                              hipStream_t stream) {
  (void)in_sizes; (void)n_in; (void)out_size; (void)d_ws; (void)ws_size;
  const _Float16* X  = (const _Float16*)d_in[0];   // [4,4096,4096] fp16
  const _Float16* Wt = (const _Float16*)d_in[1];   // [4,4096,4096] fp16
  _Float16* Out = (_Float16*)d_out;                // [4096,4096] fp16

  dim3 grid(ND / BN, MD / BM);                     // 16 x 32 tiles
  dim3 block(256);                                 // 8 wave32
  size_t shmem = (size_t)2 * (LDS_A + LDS_B) * sizeof(_Float16); // 110,592 B of 320 KB
  hipLaunchKernelGGL(rowpar_allreduce_gemm_wmma, grid, block, shmem, stream, X, Wt, Out);
}
